// BinaryLayer_70265664962797
// MI455X (gfx1250) — compile-verified, matrix operand order checked by reading, not run on hardware
//
#include <hip/hip_runtime.h>
#include <stdint.h>

// Problem constants (match reference)
#define B_N    1024          // batch
#define F_N    784           // features
#define OUT_N  1024          // outputs
#define OR_T   32            // OR terms per output
#define AND_T  16            // AND leaves per term
#define K_REAL 1569          // 1 + 2*F
#define KPAD   1600          // padded to multiple of 64 for iu8 WMMA K
#define NKB    25            // KPAD / 64 K-blocks
#define APITCH 1600          // packed-x row pitch (bytes)
#define BPITCH 1608          // LDS B-panel pitch: 8B aligned, avoids 64-bank stride conflicts
#define MT     4             // m-tiles per wave (16 rows each) -> 8 waves cover 512 rows

typedef __attribute__((ext_vector_type(8))) int v8i;

// ---------------------------------------------------------------------------
// Kernel 1: pack x into A' with a fragment-friendly swizzle.
// Logical clause vector per row: [1, x!=0, !(x!=0), 0-pad] (KPAD entries).
// Storage: A'[row][kb][half][j][b] at byte  row*1600 + kb*64 + half*32 + j*8 + b
// holding logical k = kb*64 + j*16 + half*8 + b.  With this layout, the lane
// owning (row, half) reads its whole 16x64 iu8 A-fragment slice as 32
// contiguous bytes = 2x global_load_b128, landing in VGPR order 0..7.
// ---------------------------------------------------------------------------
__global__ __launch_bounds__(256) void pack_x_kernel(const float* __restrict__ x,
                                                     uint8_t* __restrict__ A) {
    int idx = blockIdx.x * 256 + threadIdx.x;
    if (idx >= B_N * APITCH) return;
    int row = idx / APITCH;
    int r   = idx - row * APITCH;
    int kb   = r >> 6;        // 64-byte K-block
    int r2   = r & 63;
    int half = r2 >> 5;       // which lane-half owns these bytes
    int r3   = r2 & 31;
    int j    = r3 >> 3;       // fragment VGPR pair
    int b    = r3 & 7;
    int k = kb * 64 + j * 16 + half * 8 + b;   // logical clause index

    uint8_t v = 0;
    if (k == 0)                 v = 1;                                            // constant-1 column
    else if (k <= F_N)          v = (x[row * F_N + (k - 1)]       != 0.0f) ? 1 : 0; // xb
    else if (k <= 2 * F_N)      v = (x[row * F_N + (k - 1 - F_N)] != 0.0f) ? 0 : 1; // ~xb
    // k in [1569,1600) stays 0 (K padding)
    A[idx] = v;
}

// ---------------------------------------------------------------------------
// Kernel 2: grid (OUT_N, 2). blockIdx.x = output column o,
// blockIdx.y = batch half (512 rows).
//   Phase 1: build 32x1600 binary selection panel in LDS + per-term
//            distinct-count targets and all-zero masks.
//   Phase 2: iu8 WMMA GEMM  (512 x 1600) x (1600 x 32) with i32 accum,
//            8 waves x 4 m-tiles x 2 n-tiles.  K-loop FULLY UNROLLED (25
//            steps): no loop phis -> no accumulator rotation copies and no
//            WMMA->VALU hazard NOPs; loads pipeline freely across steps.
//   Phase 3: (dot == target) & mask, OR over 32 terms via wave32 ballots,
//            store float 0/1 to out[b*OUT_N + o].
// ---------------------------------------------------------------------------
__global__ __launch_bounds__(256) void tsetlin_wmma_kernel(
    const int* __restrict__ W,       // (OUT, OR_T, AND_T) int32
    const uint8_t* __restrict__ A,   // (B_N, APITCH) swizzled packed x
    float* __restrict__ out) {       // (B_N, OUT_N) float

    __shared__ uint8_t Bp[OR_T * BPITCH];   // ~50.3 KB binary B panel (N-major)
    __shared__ int s_tgt[OR_T];
    __shared__ int s_mask[OR_T];

    const int tid    = threadIdx.x;
    const int o      = blockIdx.x;
    const int mhalf  = blockIdx.y;         // which 512-row half of the batch
    const int lane   = tid & 31;
    const int wave   = tid >> 5;
    const int lane15 = lane & 15;
    const int half   = lane >> 4;          // which 8-byte K-half this lane owns

    // ---- Phase 1a: zero the LDS panel cooperatively --------------------------
    {
        uint64_t* p = (uint64_t*)Bp;
        const int n64 = (OR_T * BPITCH) / 8;
        for (int i = tid; i < n64; i += 256) p[i] = 0;
    }
    __syncthreads();

    // ---- Phase 1b: threads 0..31 each build one term's row -------------------
    if (tid < OR_T) {
        int w[AND_T];
        bool allz = true;
        const int* wp = W + ((size_t)o * OR_T + tid) * AND_T;
#pragma unroll
        for (int a = 0; a < AND_T; ++a) {
            w[a] = wp[a];
            if (w[a] != 0) allz = false;
        }
        int cnt = 0;
#pragma unroll
        for (int a = 0; a < AND_T; ++a) {
            bool dup = false;
            for (int b2 = 0; b2 < a; ++b2) dup = dup || (w[b2] == w[a]);
            if (!dup) {
                ++cnt;
                unsigned idx = (unsigned)w[a];
                if (idx < (unsigned)KPAD) Bp[tid * BPITCH + idx] = 1;
            }
        }
        s_tgt[tid]  = cnt;            // dot == cnt  <=>  all distinct bits true
        s_mask[tid] = allz ? 0 : 1;   // or_mask
    }
    __syncthreads();

    // ---- Phase 2: GEMM -------------------------------------------------------
    v8i acc0[MT], acc1[MT];
    const v8i vzero = {0, 0, 0, 0, 0, 0, 0, 0};
#pragma unroll
    for (int i = 0; i < MT; ++i) { acc0[i] = vzero; acc1[i] = vzero; }

    const int rowBase = mhalf * 512 + wave * (MT * 16) + lane15;
    const uint8_t* aBase  = A + (size_t)rowBase * APITCH + half * 32; // swizzled
    const uint8_t* b0Base = &Bp[(0 * 16 + lane15) * BPITCH + half * 8];
    const uint8_t* b1Base = &Bp[(1 * 16 + lane15) * BPITCH + half * 8];

#pragma unroll
    for (int kb = 0; kb < NKB; ++kb) {
        const int k0 = kb * 64;

        // B fragments from LDS (shared by all m-tiles of this wave)
        int2 q0 = *(const int2*)(b0Base + k0);
        int2 q1 = *(const int2*)(b0Base + k0 + 16);
        int2 q2 = *(const int2*)(b0Base + k0 + 32);
        int2 q3 = *(const int2*)(b0Base + k0 + 48);
        v8i bf0 = {q0.x, q0.y, q1.x, q1.y, q2.x, q2.y, q3.x, q3.y};

        int2 r0 = *(const int2*)(b1Base + k0);
        int2 r1 = *(const int2*)(b1Base + k0 + 16);
        int2 r2 = *(const int2*)(b1Base + k0 + 32);
        int2 r3 = *(const int2*)(b1Base + k0 + 48);
        v8i bf1 = {r0.x, r0.y, r1.x, r1.y, r2.x, r2.y, r3.x, r3.y};

        // A fragments: 32 contiguous bytes per lane = 2x b128 per m-tile.
        v8i af[MT];
#pragma unroll
        for (int i = 0; i < MT; ++i) {
            const uint8_t* ap = aBase + (size_t)i * (16 * APITCH) + k0;
            int4 lo = *(const int4*)(ap);        // VGPRs 0..3 of fragment
            int4 hi = *(const int4*)(ap + 16);   // VGPRs 4..7 of fragment
            v8i t = {lo.x, lo.y, lo.z, lo.w, hi.x, hi.y, hi.z, hi.w};
            af[i] = t;
        }

#pragma unroll
        for (int i = 0; i < MT; ++i) {
            // unsigned u8 x u8 -> i32 accumulate (in place)
            acc0[i] = __builtin_amdgcn_wmma_i32_16x16x64_iu8(
                false, af[i], false, bf0, acc0[i], false, false);
            acc1[i] = __builtin_amdgcn_wmma_i32_16x16x64_iu8(
                false, af[i], false, bf1, acc1[i], false, false);
        }
    }

    // ---- Phase 3: threshold, OR-reduce over 32 terms, store ------------------
    // C layout (i32 16x16): lane L holds N = L&15; VGPR r holds M=r (lanes 0-15)
    // and M=r+8 (lanes 16-31).
    const int tgt0 = s_tgt[lane15];
    const int tgt1 = s_tgt[16 + lane15];
    const int mk0  = s_mask[lane15];
    const int mk1  = s_mask[16 + lane15];

#pragma unroll
    for (int i = 0; i < MT; ++i) {
        unsigned rows = 0;  // bit b => OR over 32 terms for batch row mbase+b
#pragma unroll
        for (int r = 0; r < 8; ++r) {
            bool p0 = mk0 && (acc0[i][r] == tgt0);
            bool p1 = mk1 && (acc1[i][r] == tgt1);
            unsigned bb = __builtin_amdgcn_ballot_w32(p0) |
                          __builtin_amdgcn_ballot_w32(p1);
            rows |= ((bb & 0xFFFFu) ? 1u : 0u) << r;        // row r   (lanes 0-15)
            rows |= ((bb >> 16)     ? 1u : 0u) << (r + 8);  // row r+8 (lanes 16-31)
        }
        const int mbase = mhalf * 512 + wave * (MT * 16) + i * 16;
        if (lane < 16) {
            out[(size_t)(mbase + lane) * OUT_N + o] =
                ((rows >> lane) & 1u) ? 1.0f : 0.0f;
        }
    }
}

// ---------------------------------------------------------------------------
// Harness entry point
// ---------------------------------------------------------------------------
extern "C" void kernel_launch(void* const* d_in, const int* in_sizes, int n_in,
                              void* d_out, int out_size, void* d_ws, size_t ws_size,
                              hipStream_t stream) {
    const float* x = (const float*)d_in[0];   // (1024, 784) float32
    const int*   w = (const int*)d_in[1];     // (1024, 32, 16) int32
    float* out = (float*)d_out;               // (1024, 1024) float
    uint8_t* A = (uint8_t*)d_ws;              // 1024*1600 = 1.6 MB swizzled packed x

    pack_x_kernel<<<(B_N * APITCH + 255) / 256, 256, 0, stream>>>(x, A);
    dim3 grid(OUT_N, 2, 1);
    tsetlin_wmma_kernel<<<grid, 256, 0, stream>>>(w, A, out);
}
